// LTCPredictor_49581102465089
// MI455X (gfx1250) — matrix operand order repaired and checked by reading, not compiled, not run on hardware
//
#include <hip/hip_runtime.h>

typedef _Float16 v16h __attribute__((ext_vector_type(16)));
typedef _Float16 v8h  __attribute__((ext_vector_type(8)));
typedef float    v8f  __attribute__((ext_vector_type(8)));
typedef float    v4f  __attribute__((ext_vector_type(4)));

namespace {
constexpr int Bb  = 64;    // batch
constexpr int Tt  = 2048;  // timesteps
constexpr int Ii  = 128;   // input dim
constexpr int Hh  = 256;   // hidden dim
constexpr int MB  = 16;    // batch rows per block (one WMMA M-tile)
constexpr int LDH = Hh + 8; // f16 LDS row stride for h  (528B = 33*16, 16B aligned, bank-skewed)
constexpr int LDX = Ii + 8; // f16 LDS row stride for x_t (272B = 17*16)
constexpr int NTHREADS = 512; // 16 waves of 32
constexpr int UNF = 6;
constexpr float DT    = 1.0f / 6.0f;
constexpr float LOG2E = 1.4426950408889634f;
}

// Split work-group barrier: signal as soon as LDS reads retire, overlap
// register-only compute with barrier completion, then wait.
#if __has_builtin(__builtin_amdgcn_s_barrier_signal) && \
    __has_builtin(__builtin_amdgcn_s_barrier_wait) &&   \
    __has_builtin(__builtin_amdgcn_s_wait_dscnt)
#define LTC_SPLIT_BARRIER 1
#define LTC_BAR_SIGNAL()  do { __builtin_amdgcn_s_wait_dscnt(0);            \
                               __builtin_amdgcn_s_barrier_signal(-1); } while (0)
#define LTC_BAR_WAIT()    __builtin_amdgcn_s_barrier_wait(-1)
#else
#define LTC_SPLIT_BARRIER 0
#define LTC_BAR_SIGNAL()  do { } while (0)
#define LTC_BAR_WAIT()    __syncthreads()
#endif

// Load a 16x32 f16 A-fragment (row-major source in LDS) per ISA layout:
// lane L (L<16): row M=L, K = kcol..kcol+7 in V0..3, K = kcol+16..kcol+23 in V4..7
// lane L+16   : same row, kcol offset by +8 (handled by caller via kcol)
__device__ __forceinline__ v16h load_a16(const _Float16* base, int ld, int row, int kcol) {
  const _Float16* p = base + row * ld + kcol;
  v16h r;
  *(v8h*)&r       = *(const v8h*)p;         // K = kcol .. kcol+7
  *((v8h*)&r + 1) = *(const v8h*)(p + 16);  // K = kcol+16 .. kcol+23
  return r;
}

__global__ __launch_bounds__(NTHREADS, 1)
void ltc_scan_kernel(const float* __restrict__ x,      // [B, T, I]
                     const float* __restrict__ Win,    // [H, I]
                     const float* __restrict__ Wrec,   // [H, H]
                     const float* __restrict__ bias,   // [H]
                     const float* __restrict__ Avec,   // [H]
                     const float* __restrict__ tau,    // [H]
                     const float* __restrict__ head_w, // [1, H]
                     const float* __restrict__ head_b, // [1]
                     float* __restrict__ out)          // [B, 1]
{
  __shared__ _Float16 lds_h[MB][LDH];  // current hidden state, f16
  __shared__ _Float16 xbuf[MB][LDX];   // x_t staged as f16
  __shared__ float    red[16][17];     // epilogue reduction

  const int tid  = threadIdx.x;
  const int wave = tid >> 5;
  const int lane = tid & 31;
  const int n    = lane & 15;       // column within N-tile / row within M-tile
  const int hi   = lane >> 4;       // half-wave selector
  const int j    = wave * 16 + n;   // hidden column owned by this lane
  const int b0   = blockIdx.x * MB; // batch base for this block

  // Per-lane constants for column j (hoisted out of the 12288-step serial loop)
  const float dtA  = DT * Avec[j];
  const float den0 = 1.0f + DT / tau[j];
  const float bj   = bias[j];
  const float hw   = head_w[j];

  // ---- Pin W_rec as 8 B-fragments (K=256) in registers, f16 ----
  // B layout: lane n = column j; lanes 0-15 hold K=kf*32+{0..7,16..23}, lanes 16-31 hold +{8..15,24..31},
  // packed 2 K-values per VGPR in order.
  v16h wrec[8];
  #pragma unroll
  for (int kf = 0; kf < 8; ++kf) {
    const float* src = Wrec + (size_t)j * Hh + kf * 32 + hi * 8;
    #pragma unroll
    for (int e = 0; e < 16; ++e) wrec[kf][e] = (_Float16)src[e < 8 ? e : e + 8];
  }
  // ---- Pin W_in as 4 B-fragments (K=128) in registers, f16 ----
  v16h win[4];
  #pragma unroll
  for (int kf = 0; kf < 4; ++kf) {
    const float* src = Win + (size_t)j * Ii + kf * 32 + hi * 8;
    #pragma unroll
    for (int e = 0; e < 16; ++e) win[kf][e] = (_Float16)src[e < 8 ? e : e + 8];
  }

  // h state in f32 registers, C/D layout: element e -> (row = e + 8*hi, col = j)
  v8f h;
  #pragma unroll
  for (int e = 0; e < 8; ++e) h[e] = 0.0f;

  // h0 = 0 in LDS
  for (int idx = tid; idx < MB * LDH; idx += NTHREADS)
    (&lds_h[0][0])[idx] = (_Float16)0.0f;
  __syncthreads();

  // x_t staging assignment: 16 rows x 128 cols, 4 floats per thread
  const int xrow = tid >> 5;
  const int xcol = (tid & 31) * 4;
  const float* xg = x + ((size_t)(b0 + xrow) * Tt) * Ii + xcol;

  for (int t = 0; t < Tt; ++t) {
    // ---- stage x_t -> LDS (fp32 -> f16) + near-scope prefetch of t+1 ----
    {
      const float* gp = xg + (size_t)t * Ii;
      v4f v = *(const v4f*)gp;
      union { _Float16 hh[4]; unsigned long long u; } pk;
      #pragma unroll
      for (int e = 0; e < 4; ++e) pk.hh[e] = (_Float16)v[e];
      *(unsigned long long*)&xbuf[xrow][xcol] = pk.u;
      if (t + 1 < Tt) __builtin_prefetch(gp + Ii, 0, 3);
    }
    __syncthreads();

    // ---- xin = bias + x_t @ W_in^T  (frozen over the 6 unfolds) ----
    // two independent 2-WMMA chains to hide WMMA RAW latency
    v8f xin, xin1;
    #pragma unroll
    for (int e = 0; e < 8; ++e) { xin[e] = bj; xin1[e] = 0.0f; }
    #pragma unroll
    for (int kf = 0; kf < 2; ++kf) {
      v16h a0 = load_a16(&xbuf[0][0], LDX, n, kf * 32 + hi * 8);
      v16h a1 = load_a16(&xbuf[0][0], LDX, n, (kf + 2) * 32 + hi * 8);
      xin  = __builtin_amdgcn_wmma_f32_16x16x32_f16(
          false, a0, false, win[kf], (short)0, xin, false, false);
      xin1 = __builtin_amdgcn_wmma_f32_16x16x32_f16(
          false, a1, false, win[kf + 2], (short)0, xin1, false, false);
    }
    #pragma unroll
    for (int e = 0; e < 8; ++e) xin[e] += xin1[e];

    // ---- 6 fused semi-implicit Euler sub-steps ----
    #pragma unroll 1
    for (int u = 0; u < UNF; ++u) {
      // s = xin + h @ W_rec^T : K=256 as FOUR independent depth-2 WMMA chains,
      // round-robined so each chain's RAW dependency is covered by the other
      // three chains' issue slots (f16 WMMA->WMMA hazard is ~5 slots).
      v8f s0 = xin;
      v8f s1, s2, s3;
      #pragma unroll
      for (int e = 0; e < 8; ++e) { s1[e] = 0.0f; s2[e] = 0.0f; s3[e] = 0.0f; }
      #pragma unroll
      for (int kf = 0; kf < 2; ++kf) {
        v16h a0 = load_a16(&lds_h[0][0], LDH, n, (kf + 0) * 32 + hi * 8);
        v16h a1 = load_a16(&lds_h[0][0], LDH, n, (kf + 2) * 32 + hi * 8);
        v16h a2 = load_a16(&lds_h[0][0], LDH, n, (kf + 4) * 32 + hi * 8);
        v16h a3 = load_a16(&lds_h[0][0], LDH, n, (kf + 6) * 32 + hi * 8);
        s0 = __builtin_amdgcn_wmma_f32_16x16x32_f16(
            false, a0, false, wrec[kf + 0], (short)0, s0, false, false);
        s1 = __builtin_amdgcn_wmma_f32_16x16x32_f16(
            false, a1, false, wrec[kf + 2], (short)0, s1, false, false);
        s2 = __builtin_amdgcn_wmma_f32_16x16x32_f16(
            false, a2, false, wrec[kf + 4], (short)0, s2, false, false);
        s3 = __builtin_amdgcn_wmma_f32_16x16x32_f16(
            false, a3, false, wrec[kf + 6], (short)0, s3, false, false);
      }

      // Announce "done reading old h" now; overlap the register-only
      // elementwise update with barrier completion across waves.
      LTC_BAR_SIGNAL();

      // elementwise: f = sigmoid(s); h = (h + DT*A*f) / (1 + DT/tau + DT*f)
      #pragma unroll
      for (int e = 0; e < 8; ++e) {
        float s = (s0[e] + s2[e]) + (s1[e] + s3[e]);
        float f = __builtin_amdgcn_rcpf(
            1.0f + __builtin_amdgcn_exp2f(-LOG2E * s));
        h[e] = (h[e] + dtA * f) * __builtin_amdgcn_rcpf(den0 + DT * f);
      }

      LTC_BAR_WAIT();   // all waves done reading old h
      #pragma unroll
      for (int e = 0; e < 8; ++e)
        lds_h[e + hi * 8][j] = (_Float16)h[e];
      LTC_BAR_SIGNAL(); // my stores retired (dscnt==0) -> announce
      LTC_BAR_WAIT();   // everyone's new h visible
#if !LTC_SPLIT_BARRIER
      __syncthreads();  // fallback needs the second full barrier
#endif
    }
  }

  // ---- head: out[b] = sum_j h[b,j] * head_w[j] + head_b ----
  #pragma unroll
  for (int e = 0; e < 8; ++e) {
    float p = h[e] * hw;
    p += __shfl_xor(p, 1, 32);
    p += __shfl_xor(p, 2, 32);
    p += __shfl_xor(p, 4, 32);
    p += __shfl_xor(p, 8, 32);  // sum over the 16 j's in each half-wave
    if (n == 0) red[wave][e + hi * 8] = p;
  }
  __syncthreads();
  if (tid < MB) {
    float acc = head_b[0];
    #pragma unroll
    for (int w = 0; w < 16; ++w) acc += red[w][tid];
    out[b0 + tid] = acc;
  }
}

extern "C" void kernel_launch(void* const* d_in, const int* in_sizes, int n_in,
                              void* d_out, int out_size, void* d_ws, size_t ws_size,
                              hipStream_t stream) {
  (void)in_sizes; (void)n_in; (void)d_ws; (void)ws_size; (void)out_size;
  const float* x      = (const float*)d_in[0];
  const float* Win    = (const float*)d_in[1];
  const float* Wrec   = (const float*)d_in[2];
  const float* bias   = (const float*)d_in[3];
  const float* Avec   = (const float*)d_in[4];
  const float* tau    = (const float*)d_in[5];
  const float* head_w = (const float*)d_in[6];
  const float* head_b = (const float*)d_in[7];
  float* out = (float*)d_out;

  ltc_scan_kernel<<<dim3(Bb / MB), dim3(NTHREADS), 0, stream>>>(
      x, Win, Wrec, bias, Avec, tau, head_w, head_b, out);
}
